// PositionEvaluator_76218489635166
// MI455X (gfx1250) — matrix operand order repaired and checked by reading, not compiled, not run on hardware
//
#include <hip/hip_runtime.h>

// ---------------------------------------------------------------------------
// CDNA5 (gfx1250) fused gated-vector-perceptron chain for PositionEvaluator.
// One workgroup (4 wave32) processes a tile of 16 focal points; every matvec
// in the layer chain becomes v_wmma_f32_16x16x32_f16 tiles (f16 in, f32 acc).
// Intermediates are staged in LDS using the documented WMMA VGPR layouts.
// ---------------------------------------------------------------------------

typedef _Float16 v8h  __attribute__((ext_vector_type(8)));
typedef _Float16 v16h __attribute__((ext_vector_type(16)));
typedef float    v8f  __attribute__((ext_vector_type(8)));

#define TPB   128      // 4 wave32
#define LDV   264      // vector-feature LDS row (max 256 cols + 8 pad)
#define LDC   392      // concat buffer row (max 384 + 8 pad)
#define LDQ   264      // scalar-input row (max 256 + 8 pad)
#define LDSS  136      // out_s row (128 + 8 pad)

__device__ __forceinline__ v8f wmma_f16(v16h a, v16h b, v8f c) {
  return __builtin_amdgcn_wmma_f32_16x16x32_f16(false, a, false, b,
                                                (short)0, c, false, false);
}

// A fragment (16x32 f16): lane m=L&15, half=L>>4.
// a[0..7]  = row[half*8 .. half*8+7], a[8..15] = row[16+half*8 .. +7]
__device__ __forceinline__ v16h loadA(const _Float16* row, int half) {
  v8h lo = *(const v8h*)(row + half * 8);
  v8h hi = *(const v8h*)(row + 16 + half * 8);
  v16h a;
#pragma unroll
  for (int j = 0; j < 8; ++j) { a[j] = lo[j]; a[j + 8] = hi[j]; }
  return a;
}

// B fragment (32x16 f16): lane n=L&15, half=L>>4; b[0..7] = wrow[half*8 .. +7], b[8..15] = wrow[16+half*8 .. +7]
__device__ __forceinline__ v16h loadB(const _Float16* row, int half) {
  v8h lo = *(const v8h*)(row + half * 8);
  v8h hi = *(const v8h*)(row + 16 + half * 8);
  v16h b;
#pragma unroll
  for (int j = 0; j < 8; ++j) { b[j] = lo[j]; b[j + 8] = hi[j]; }
  return b;
}

// C/D (16x16 f32): element (vgpr j, lane L) = OUT[j + 8*(L>>4)][L&15]
__device__ __forceinline__ void storeC(v8f c, _Float16* D, int ld, int n0, int lane) {
  const int n = lane & 15, half = lane >> 4;
#pragma unroll
  for (int j = 0; j < 8; ++j)
    D[(j + 8 * half) * ld + n0 + n] = (_Float16)c[j];
}

// OUT[16][16 tile @ n0] = A[16][K] * W[N][K]^T   (A in LDS, W f16 in global)
__device__ __forceinline__ v8f gemm1(const _Float16* A, int lda,
                                     const _Float16* W, int K, int n0, int lane) {
  const int m = lane & 15, half = lane >> 4;
  const _Float16* ar = A + m * lda;
  const _Float16* wr = W + (size_t)(n0 + m) * K;
  v8f c = {0.f, 0.f, 0.f, 0.f, 0.f, 0.f, 0.f, 0.f};
  for (int k0 = 0; k0 < K; k0 += 32)
    c = wmma_f16(loadA(ar + k0, half), loadB(wr + k0, half), c);
  return c;
}

// Three GEMMs (x/y/z channels) sharing one B fragment per K-step.
__device__ __forceinline__ void gemm3(const _Float16* Ax, const _Float16* Ay,
                                      const _Float16* Az, int lda,
                                      const _Float16* W, int K, int n0, int lane,
                                      v8f& cx, v8f& cy, v8f& cz) {
  const int m = lane & 15, half = lane >> 4;
  const _Float16* arx = Ax + m * lda;
  const _Float16* ary = Ay + m * lda;
  const _Float16* arz = Az + m * lda;
  const _Float16* wr  = W + (size_t)(n0 + m) * K;
  v8f z = {0.f, 0.f, 0.f, 0.f, 0.f, 0.f, 0.f, 0.f};
  cx = z; cy = z; cz = z;
  for (int k0 = 0; k0 < K; k0 += 32) {
    v16h b = loadB(wr + k0, half);
    cx = wmma_f16(loadA(arx + k0, half), b, cx);
    cy = wmma_f16(loadA(ary + k0, half), b, cy);
    cz = wmma_f16(loadA(arz + k0, half), b, cz);
  }
}

struct LW {
  const _Float16 *Wvh, *Ws, *Wvo, *Wg, *Wd;
  const float* bg;
  int Cin, H, Sin, Sout, O, act;
};

// One gated-vector layer for a 16-focal tile (4 waves cooperate).
__device__ __forceinline__ void gv_layer(
    int tid, const LW L,
    const _Float16* viX, const _Float16* viY, const _Float16* viZ,  // input vec
    _Float16* vhX, _Float16* vhY, _Float16* vhZ,                    // vh scratch
    _Float16* voX, _Float16* voY, _Float16* voZ,                    // output vec
    _Float16* vdX, _Float16* vdY, _Float16* vdZ,                    // d temp (may alias input)
    _Float16* sCat, _Float16* sS, _Float16* sIn) {
  const int lane = tid & 31;
  const int wv   = tid >> 5;

  // (1) scalar features -> concat tail
  for (int e = tid; e < 16 * L.Sin; e += TPB) {
    int r = e / L.Sin, c = e - r * L.Sin;
    sCat[r * LDC + L.H + c] = sIn[r * LDQ + c];
  }
  // (2) vh = V * Wvh^T ; vnorm -> concat head
  for (int t = wv; t < (L.H >> 4); t += 4) {
    int n0 = t << 4;
    v8f cx, cy, cz;
    gemm3(viX, viY, viZ, LDV, L.Wvh, L.Cin, n0, lane, cx, cy, cz);
    storeC(cx, vhX, LDV, n0, lane);
    storeC(cy, vhY, LDV, n0, lane);
    storeC(cz, vhZ, LDV, n0, lane);
    v8f vn;
#pragma unroll
    for (int j = 0; j < 8; ++j)
      vn[j] = sqrtf(cx[j] * cx[j] + cy[j] * cy[j] + cz[j] * cz[j]);
    storeC(vn, sCat, LDC, n0, lane);
  }
  __syncthreads();
  // (3) out_s = concat * Ws^T
  for (int t = wv; t < (L.Sout >> 4); t += 4) {
    int n0 = t << 4;
    v8f c = gemm1(sCat, LDC, L.Ws, L.H + L.Sin, n0, lane);
    storeC(c, sS, LDSS, n0, lane);
  }
  __syncthreads();
  // (4) out_v = sigmoid(out_s*Wg^T + bg) * (vh*Wvo^T)
  for (int t = wv; t < (L.O >> 4); t += 4) {
    int n0 = t << 4;
    v8f g = gemm1(sS, LDSS, L.Wg, L.Sout, n0, lane);
    v8f cx, cy, cz;
    gemm3(vhX, vhY, vhZ, LDV, L.Wvo, L.H, n0, lane, cx, cy, cz);
    float bgv = L.bg[n0 + (lane & 15)];
#pragma unroll
    for (int j = 0; j < 8; ++j) {
      float gg = 1.0f / (1.0f + __expf(-(g[j] + bgv)));
      cx[j] *= gg; cy[j] *= gg; cz[j] *= gg;
    }
    storeC(cx, voX, LDV, n0, lane);
    storeC(cy, voY, LDV, n0, lane);
    storeC(cz, voZ, LDV, n0, lane);
  }
  if (L.act) {
    // (5a) scalar leaky-relu -> next-layer scalar input
    for (int e = tid; e < 16 * L.Sout; e += TPB) {
      int r = e / L.Sout, c = e - r * L.Sout;
      float v = (float)sS[r * LDSS + c];
      sIn[r * LDQ + c] = (_Float16)(v > 0.f ? v : 0.01f * v);
    }
    __syncthreads();
    // (5b) d = out_v * Wd^T  (into dead input set)
    for (int t = wv; t < (L.O >> 4); t += 4) {
      int n0 = t << 4;
      v8f dx, dy, dz;
      gemm3(voX, voY, voZ, LDV, L.Wd, L.O, n0, lane, dx, dy, dz);
      storeC(dx, vdX, LDV, n0, lane);
      storeC(dy, vdY, LDV, n0, lane);
      storeC(dz, vdZ, LDV, n0, lane);
    }
    __syncthreads();
    // (5c) vector leaky-relu, in place on out_v
    for (int e = tid; e < 16 * L.O; e += TPB) {
      int r = e / L.O, c = e - r * L.O;
      float x0 = (float)voX[r * LDV + c], x1 = (float)voY[r * LDV + c], x2 = (float)voZ[r * LDV + c];
      float d0 = (float)vdX[r * LDV + c], d1 = (float)vdY[r * LDV + c], d2 = (float)vdZ[r * LDV + c];
      float dot = x0 * d0 + x1 * d1 + x2 * d2;
      float k = dot / (d0 * d0 + d1 * d1 + d2 * d2 + 1e-6f);
      float o0, o1, o2;
      if (dot >= 0.f) { o0 = x0; o1 = x1; o2 = x2; }
      else { o0 = x0 - k * d0; o1 = x1 - k * d1; o2 = x2 - k * d2; }
      voX[r * LDV + c] = (_Float16)(0.2f * x0 + 0.8f * o0);
      voY[r * LDV + c] = (_Float16)(0.2f * x1 + 0.8f * o1);
      voZ[r * LDV + c] = (_Float16)(0.2f * x2 + 0.8f * o2);
    }
    __syncthreads();
  } else {
    for (int e = tid; e < 16 * L.Sout; e += TPB) {
      int r = e / L.Sout, c = e - r * L.Sout;
      sIn[r * LDQ + c] = sS[r * LDSS + c];
    }
    __syncthreads();
  }
}

struct Params {
  const float *h_sca, *h_vec, *pos_compose, *pos;
  const int*  idx_focal;
  float* out;
  const _Float16 *g1Wvh, *g1Ws, *g1Wvo, *g1Wg, *g1Wd;
  const _Float16 *g2Wvh, *g2Ws, *g2Wvo, *g2Wg;
  const _Float16 *a1Wvh, *a1Ws, *a1Wvo, *a1Wg, *a1Wd;
  const _Float16 *a2Wvh, *a2Ws, *a2Wvo, *a2Wg, *a2Wd;
  const _Float16 *fWvh;
  const float *fWs;
  const float *g1bg, *g2bg, *a1bg, *a2bg;
  int F;
};

#define SMEM_BYTES (320 + 2 * (9 * 16 * LDV + 16 * LDC + 16 * LDQ + 16 * LDSS))

__global__ __launch_bounds__(TPB) void pe_main(Params P) {
  extern __shared__ char smem[];
  float* sRel = (float*)smem;            // 16 x 4 f32
  int*   sIdx = (int*)(smem + 256);      // 16 ints
  _Float16* fp = (_Float16*)(smem + 320);
  _Float16 *V0x = fp; fp += 16 * LDV;
  _Float16 *V0y = fp; fp += 16 * LDV;
  _Float16 *V0z = fp; fp += 16 * LDV;
  _Float16 *VHx = fp; fp += 16 * LDV;
  _Float16 *VHy = fp; fp += 16 * LDV;
  _Float16 *VHz = fp; fp += 16 * LDV;
  _Float16 *V1x = fp; fp += 16 * LDV;
  _Float16 *V1y = fp; fp += 16 * LDV;
  _Float16 *V1z = fp; fp += 16 * LDV;
  _Float16 *sCat = fp; fp += 16 * LDC;
  _Float16 *sIn  = fp; fp += 16 * LDQ;
  _Float16 *sS   = fp; fp += 16 * LDSS;

  const int tid  = threadIdx.x;
  const int lane = tid & 31;
  const int base = blockIdx.x * 16;

  // ---- gather phase ----
  if (tid < 16) {
    int g = base + tid; if (g >= P.F) g = P.F - 1;
    int idx = P.idx_focal[g];
    sIdx[tid] = idx;
    sRel[tid * 4 + 0] = P.pos[(size_t)g * 3 + 0] - P.pos_compose[(size_t)idx * 3 + 0];
    sRel[tid * 4 + 1] = P.pos[(size_t)g * 3 + 1] - P.pos_compose[(size_t)idx * 3 + 1];
    sRel[tid * 4 + 2] = P.pos[(size_t)g * 3 + 2] - P.pos_compose[(size_t)idx * 3 + 2];
  }
  __syncthreads();
  for (int e = tid; e < 16 * 256; e += TPB) {           // h_sca gather -> f16
    int r = e >> 8, c = e & 255;
    sIn[r * LDQ + c] = (_Float16)P.h_sca[(size_t)sIdx[r] * 256 + c];
  }
  for (int e = tid; e < 16 * 64; e += TPB) {            // h_vec gather, split xyz
    int r = e >> 6, c = e & 63;
    const float* p = P.h_vec + (size_t)sIdx[r] * 192 + c * 3;
    V0x[r * LDV + c] = (_Float16)p[0];
    V0y[r * LDV + c] = (_Float16)p[1];
    V0z[r * LDV + c] = (_Float16)p[2];
  }
  __syncthreads();

  // ---- layer chain ----
  { LW L = {P.g1Wvh, P.g1Ws, P.g1Wvo, P.g1Wg, P.g1Wd, P.g1bg, 64, 128, 256, 128, 128, 1};
    gv_layer(tid, L, V0x, V0y, V0z, VHx, VHy, VHz, V1x, V1y, V1z, V0x, V0y, V0z, sCat, sS, sIn); }
  { LW L = {P.g2Wvh, P.g2Ws, P.g2Wvo, P.g2Wg, (const _Float16*)0, P.g2bg, 128, 128, 128, 128, 128, 0};
    gv_layer(tid, L, V1x, V1y, V1z, VHx, VHy, VHz, V0x, V0y, V0z, V1x, V1y, V1z, sCat, sS, sIn); }
  { LW L = {P.a1Wvh, P.a1Ws, P.a1Wvo, P.a1Wg, P.a1Wd, P.a1bg, 128, 256, 128, 128, 256, 1};
    gv_layer(tid, L, V0x, V0y, V0z, VHx, VHy, VHz, V1x, V1y, V1z, V0x, V0y, V0z, sCat, sS, sIn); }

  // ---- inner = <x_vec2, relpos> ; s2 = [leaky(s1) | inner] ----
  for (int e = tid; e < 16 * 128; e += TPB) {
    int r = e >> 7, c = e & 127;
    float ip = (float)V1x[r * LDV + 128 + c] * sRel[r * 4 + 0]
             + (float)V1y[r * LDV + 128 + c] * sRel[r * 4 + 1]
             + (float)V1z[r * LDV + 128 + c] * sRel[r * 4 + 2];
    sIn[r * LDQ + 128 + c] = (_Float16)ip;
  }
  __syncthreads();

  { LW L = {P.a2Wvh, P.a2Ws, P.a2Wvo, P.a2Wg, P.a2Wd, P.a2bg, 128, 128, 256, 128, 128, 1};
    gv_layer(tid, L, V1x, V1y, V1z, VHx, VHy, VHz, V0x, V0y, V0z, V1x, V1y, V1z, sCat, sS, sIn); }

  // ---- final layer: vnorm(f_Wvh) || s3, then 256-dot with f_Ws ----
  for (int e = tid; e < 16 * 128; e += TPB) {
    int r = e >> 7, c = e & 127;
    sCat[r * LDC + 128 + c] = sIn[r * LDQ + c];
  }
  {
    const int wv = tid >> 5;
    for (int t = wv; t < 8; t += 4) {
      int n0 = t << 4;
      v8f cx, cy, cz;
      gemm3(V0x, V0y, V0z, LDV, P.fWvh, 128, n0, lane, cx, cy, cz);
      v8f vn;
#pragma unroll
      for (int j = 0; j < 8; ++j)
        vn[j] = sqrtf(cx[j] * cx[j] + cy[j] * cy[j] + cz[j] * cz[j]);
      storeC(vn, sCat, LDC, n0, lane);
    }
  }
  __syncthreads();
  if (tid < 16) {
    int g = base + tid;
    if (g < P.F) {
      float acc = 0.f;
      for (int c = 0; c < 256; ++c)
        acc += (float)sCat[tid * LDC + c] * P.fWs[c];
      P.out[g] = acc;
    }
  }
}

__global__ void cvt_f16(const float* __restrict__ src, _Float16* __restrict__ dst, int n) {
  int i = blockIdx.x * 256 + threadIdx.x;
  if (i < n) dst[i] = (_Float16)src[i];
}

extern "C" void kernel_launch(void* const* d_in, const int* in_sizes, int n_in,
                              void* d_out, int out_size, void* d_ws, size_t ws_size,
                              hipStream_t stream) {
  (void)n_in; (void)out_size; (void)ws_size;
  static const int widx[20] = {4, 5, 6, 7, 9, 10, 11, 12, 13, 15,
                               16, 17, 18, 20, 21, 22, 23, 24, 26, 27};
  _Float16* w16 = (_Float16*)d_ws;
  size_t off[32] = {0};
  size_t cur = 0;
  for (int j = 0; j < 20; ++j) { off[widx[j]] = cur; cur += (size_t)in_sizes[widx[j]]; }
  for (int j = 0; j < 20; ++j) {
    int i = widx[j]; int n = in_sizes[i];
    cvt_f16<<<dim3((n + 255) / 256), dim3(256), 0, stream>>>((const float*)d_in[i], w16 + off[i], n);
  }

  Params P;
  P.h_sca = (const float*)d_in[0];
  P.h_vec = (const float*)d_in[1];
  P.pos_compose = (const float*)d_in[2];
  P.pos = (const float*)d_in[3];
  P.idx_focal = (const int*)d_in[29];
  P.out = (float*)d_out;
  P.g1Wvh = w16 + off[4];  P.g1Ws = w16 + off[5];  P.g1Wvo = w16 + off[6];
  P.g1Wg  = w16 + off[7];  P.g1Wd = w16 + off[9];
  P.g2Wvh = w16 + off[10]; P.g2Ws = w16 + off[11]; P.g2Wvo = w16 + off[12];
  P.g2Wg  = w16 + off[13];
  P.a1Wvh = w16 + off[15]; P.a1Ws = w16 + off[16]; P.a1Wvo = w16 + off[17];
  P.a1Wg  = w16 + off[18]; P.a1Wd = w16 + off[20];
  P.a2Wvh = w16 + off[21]; P.a2Ws = w16 + off[22]; P.a2Wvo = w16 + off[23];
  P.a2Wg  = w16 + off[24]; P.a2Wd = w16 + off[26];
  P.fWvh  = w16 + off[27];
  P.fWs   = (const float*)d_in[28];
  P.g1bg = (const float*)d_in[8];
  P.g2bg = (const float*)d_in[14];
  P.a1bg = (const float*)d_in[19];
  P.a2bg = (const float*)d_in[25];
  int F = in_sizes[3] / 3;   // pos is (F,3)
  P.F = F;

  int smem = SMEM_BYTES;     // ~101.7 KB per workgroup (dynamic LDS, 320KB WGP budget)
  (void)hipFuncSetAttribute(reinterpret_cast<const void*>(pe_main),
                            hipFuncAttributeMaxDynamicSharedMemorySize, smem);
  pe_main<<<dim3((F + 15) / 16), dim3(TPB), smem, stream>>>(P);
}
